// VarianceAdaptor_7997229105427
// MI455X (gfx1250) — matrix operand order repaired and checked by reading, not compiled, not run on hardware
//
#include <hip/hip_runtime.h>
#include <hip/hip_bf16.h>

typedef __attribute__((ext_vector_type(16))) _Float16 v16h;
typedef __attribute__((ext_vector_type(4)))  _Float16 v4h;
typedef __attribute__((ext_vector_type(8)))  float    v8f;

#define BATCH 16
#define LSEQ  512
#define HDIM  256
#define FDIM  256
#define TMAX  2048

// d_out layout (floats): out | pitch | energy | log_dur | mel_len
#define OUT_OFF 0
#define P_OFF   (BATCH*TMAX*HDIM)              // 8388608
#define E_OFF   (P_OFF + BATCH*LSEQ)           // 8396800
#define D_OFF   (E_OFF + BATCH*LSEQ)           // 8404992
#define ML_OFF  (D_OFF + BATCH*LSEQ)           // 8413184

#define WELEMS  (768 * 256)                    // elements per conv weight tensor

// ---------------------------------------------------------------------------
// Pre-swizzle one conv weight tensor (768 x 256 f32) into f16 WMMA B-fragment
// order: [step 0..23][wave 0..15][lane 0..31][e 0..15], where
//   K_local = (lane>>4)*16 + e,  kk = step*32 + K_local,  n = wave*16 + (lane&15)
// Grid: 384 blocks (step*16+wave) x 32 threads (lane).
// ---------------------------------------------------------------------------
__global__ __launch_bounds__(32)
void va_swizzle_w(const float* __restrict__ w, _Float16* __restrict__ out)
{
    const int step = blockIdx.x >> 4;
    const int wave = blockIdx.x & 15;
    const int lane = threadIdx.x;
    const int n    = (wave << 4) + (lane & 15);
    const int kk0  = step * 32 + ((lane >> 4) << 4);
    _Float16* o = out + (((blockIdx.x) * 32) + lane) * 16;
    #pragma unroll
    for (int e = 0; e < 16; ++e)
        o[e] = (_Float16)w[(kk0 + e) * 256 + n];
}

// ---------------------------------------------------------------------------
// Fused conv1d(k=3, C=256 -> F=256) + bias + ReLU + LayerNorm, per 16-row tile.
// Grid: B * (L/16) = 512 blocks, 512 threads (16 waves).
// Wave w computes the 16x16 output tile at columns [16w, 16w+16).
// Weights pre-swizzled to f16 fragment order (see va_swizzle_w).
// ---------------------------------------------------------------------------
__global__ __launch_bounds__(512)
void va_conv_ln(const float* __restrict__ in,       // (B, L, 256) f32
                const _Float16* __restrict__ wsw,   // swizzled (24,16,32,16) f16
                const float* __restrict__ bias,     // (256,)
                const float* __restrict__ gamma,    // (256,)
                const float* __restrict__ beta,     // (256,)
                float* __restrict__ out)            // (B, L, 256)
{
    __shared__ _Float16 As[18 * 256];   // halo rows l0-1 .. l0+16, f16
    __shared__ float    Cs[16 * 256];   // f32 tile for LN epilogue

    const int tid  = threadIdx.x;
    const int wave = tid >> 5;
    const int lane = tid & 31;
    const int m    = lane & 15;
    const int half = lane >> 4;

    const int b  = blockIdx.x >> 5;
    const int l0 = (blockIdx.x & 31) << 4;

    // Stage A tile (18 rows x 256 ch) with zero-padded halo: float4 loads,
    // packed f16 stores (cvt_pk + ds_store_b64).
    const float4* in4 = (const float4*)in;
    for (int i = tid; i < 18 * 64; i += 512) {
        const int r = i >> 6, c4 = i & 63;
        const int l = l0 - 1 + r;
        float4 v = make_float4(0.f, 0.f, 0.f, 0.f);
        if (l >= 0 && l < LSEQ) v = in4[(b * LSEQ + l) * 64 + c4];
        v4h h; h[0] = (_Float16)v.x; h[1] = (_Float16)v.y;
               h[2] = (_Float16)v.z; h[3] = (_Float16)v.w;
        *(v4h*)(&As[r * 256 + (c4 << 2)]) = h;
    }
    __syncthreads();

    const int n = (wave << 4) + m;   // global output column of this lane
    const _Float16* bfrag = wsw + ((wave << 5) + lane) * 16;  // + step*8192
    v8f acc = {};

    // K = 768 = 3 taps x 256 channels, 24 WMMA steps, fully unrolled
    #pragma unroll
    for (int step = 0; step < 24; ++step) {
        const int k  = step >> 3;          // conv tap 0..2
        const int c0 = (step & 7) << 5;    // channel base within tap

        // A fragment: contiguous k order (K_local = half*16 + e), matching swizzled B
        const _Float16* arow = &As[(m + k) * 256 + c0 + (half << 4)];
        v16h a;
        #pragma unroll
        for (int i = 0; i < 8; ++i) { a[i] = arow[i]; a[8 + i] = arow[8 + i]; }

        // B fragment: one aligned 32-byte load from pre-swizzled weights
        const v16h bf = *(const v16h*)(bfrag + step * (16 * 32 * 16));

        acc = __builtin_amdgcn_wmma_f32_16x16x32_f16(
            false, a, false, bf, (short)0, acc, false, false);
    }

    // bias + ReLU; scatter D (M = r + 8*half, N = n) into LDS f32 tile
    const float bv = bias[n];
    #pragma unroll
    for (int r = 0; r < 8; ++r) {
        float t = acc[r] + bv;
        t = t > 0.0f ? t : 0.0f;
        Cs[(r + (half << 3)) * 256 + n] = t;
    }
    __syncthreads();

    // LayerNorm over F=256: wave `wave` owns row `wave`
    const int row = wave;
    float vals[8], s = 0.0f, q = 0.0f;
    #pragma unroll
    for (int j = 0; j < 8; ++j) {
        const float v = Cs[row * 256 + lane + (j << 5)];
        vals[j] = v; s += v; q += v * v;
    }
    #pragma unroll
    for (int off = 16; off >= 1; off >>= 1) {
        s += __shfl_xor(s, off, 32);
        q += __shfl_xor(q, off, 32);
    }
    const float mu  = s * (1.0f / 256.0f);
    const float var = q * (1.0f / 256.0f) - mu * mu;
    const float rs  = rsqrtf(var + 1e-5f);

    const int gr = (b * LSEQ + l0 + row) * 256;
    #pragma unroll
    for (int j = 0; j < 8; ++j) {
        const int c = lane + (j << 5);
        out[gr + c] = (vals[j] - mu) * rs * gamma[c] + beta[c];
    }
}

// ---------------------------------------------------------------------------
// Linear head (F=256 -> 1) + mask. Grid: 1024 blocks x 256 (wave per row).
// ---------------------------------------------------------------------------
__global__ __launch_bounds__(256)
void va_linear_mask(const float* __restrict__ y,   // (B*L, 256)
                    const float* __restrict__ lw,  // (256,)
                    const float* __restrict__ lb,  // (1,)
                    const unsigned char* __restrict__ mask, // (B*L,)
                    float* __restrict__ pred)      // (B*L,)
{
    const int lane = threadIdx.x & 31;
    const int row  = blockIdx.x * 8 + (threadIdx.x >> 5);
    float s = 0.0f;
    #pragma unroll
    for (int j = 0; j < 8; ++j) {
        const int c = lane + (j << 5);
        s += y[row * 256 + c] * lw[c];
    }
    #pragma unroll
    for (int off = 16; off >= 1; off >>= 1) s += __shfl_xor(s, off, 32);
    if (lane == 0) pred[row] = mask[row] ? 0.0f : (s + lb[0]);
}

// ---------------------------------------------------------------------------
// x += emb[searchsorted(bins, target, 'right')]. Grid: B*L blocks x 256.
// ---------------------------------------------------------------------------
__global__ __launch_bounds__(256)
void va_emb_add(const float* __restrict__ in,     // (B*L, 256)
                const float* __restrict__ target, // (B*L,)
                const float* __restrict__ bins,   // (255,)
                const float* __restrict__ emb,    // (256, 256)
                float* __restrict__ out)          // (B*L, 256)
{
    __shared__ int sidx;
    const int row = blockIdx.x;
    if (threadIdx.x == 0) {
        const float v = target[row];
        int lo = 0, hi = 255;                 // insertion point in [0,255]
        while (lo < hi) {
            const int mid = (lo + hi) >> 1;
            if (bins[mid] <= v) lo = mid + 1; else hi = mid;
        }
        sidx = lo;
    }
    __syncthreads();
    const int idx = sidx;
    out[row * 256 + threadIdx.x] =
        in[row * 256 + threadIdx.x] + emb[idx * 256 + threadIdx.x];
}

// ---------------------------------------------------------------------------
// Inclusive scan of duration per batch row; emits cum + mel_len.
// Grid: B blocks x 512 threads.
// ---------------------------------------------------------------------------
__global__ __launch_bounds__(512)
void va_cumsum(const int* __restrict__ dur, int* __restrict__ cum,
               float* __restrict__ mel_len)
{
    __shared__ int s[LSEQ];
    const int b = blockIdx.x, tid = threadIdx.x;
    s[tid] = dur[b * LSEQ + tid];
    __syncthreads();
    for (int off = 1; off < LSEQ; off <<= 1) {
        int v = (tid >= off) ? s[tid - off] : 0;
        __syncthreads();
        s[tid] += v;
        __syncthreads();
    }
    cum[b * LSEQ + tid] = s[tid];
    if (tid == LSEQ - 1) mel_len[b] = (float)s[tid];
}

// ---------------------------------------------------------------------------
// Length regulation: out[b,t,:] = t < total ? x[b, idx(t), :] : 0
// Grid: B*T blocks x 256 threads.
// ---------------------------------------------------------------------------
__global__ __launch_bounds__(256)
void va_length_reg(const float* __restrict__ x,   // (B, L, 256)
                   const int* __restrict__ cum,   // (B, L)
                   float* __restrict__ out)       // (B, T, 256)
{
    __shared__ int sidx, svalid;
    const int b = blockIdx.x >> 11;
    const int t = blockIdx.x & (TMAX - 1);
    if (threadIdx.x == 0) {
        const int total = cum[b * LSEQ + (LSEQ - 1)];
        if (t >= total) { svalid = 0; sidx = 0; }
        else {
            int lo = 0, hi = LSEQ;            // searchsorted(cum, t, 'right')
            while (lo < hi) {
                const int mid = (lo + hi) >> 1;
                if (cum[b * LSEQ + mid] <= t) lo = mid + 1; else hi = mid;
            }
            sidx = lo < (LSEQ - 1) ? lo : (LSEQ - 1);
            svalid = 1;
        }
    }
    __syncthreads();
    const float v = svalid ? x[(b * LSEQ + sidx) * 256 + threadIdx.x] : 0.0f;
    out[(b * TMAX + t) * 256 + threadIdx.x] = v;
}

// ---------------------------------------------------------------------------
extern "C" void kernel_launch(void* const* d_in, const int* in_sizes, int n_in,
                              void* d_out, int out_size, void* d_ws, size_t ws_size,
                              hipStream_t stream) {
    (void)in_sizes; (void)n_in; (void)out_size; (void)ws_size;

    const float*         x      = (const float*)d_in[0];
    const unsigned char* mask   = (const unsigned char*)d_in[1];
    const float*         pitchT = (const float*)d_in[2];
    const float*         energT = (const float*)d_in[3];
    const int*           dur    = (const int*)d_in[4];
    // d_in[5] = max_len scalar (2048), compiled in.

    // predictor param tuples are flattened: 10 arrays each
    const float* const* dp = (const float* const*)&d_in[6];
    const float* const* pp = (const float* const*)&d_in[16];
    const float* const* ep = (const float* const*)&d_in[26];
    const float* pbins = (const float*)d_in[36];
    const float* ebins = (const float*)d_in[37];
    const float* pemb  = (const float*)d_in[38];
    const float* eemb  = (const float*)d_in[39];

    float* outp = (float*)d_out;

    const size_t NBL = (size_t)BATCH * LSEQ * FDIM;  // 2,097,152 floats
    float* y1  = (float*)d_ws;
    float* y2  = y1 + NBL;
    float* xe  = y2 + NBL;
    int*   cum = (int*)(xe + NBL);
    _Float16* wswBase = (_Float16*)(cum + BATCH * LSEQ);  // 6 x WELEMS halfs
    _Float16* w_dp1 = wswBase + 0 * WELEMS;
    _Float16* w_dp2 = wswBase + 1 * WELEMS;
    _Float16* w_pp1 = wswBase + 2 * WELEMS;
    _Float16* w_pp2 = wswBase + 3 * WELEMS;
    _Float16* w_ep1 = wswBase + 4 * WELEMS;
    _Float16* w_ep2 = wswBase + 5 * WELEMS;

    const dim3 gSwz(384), bSwz(32);
    const dim3 gConv(512), bConv(512);
    const dim3 gLin(1024), bLin(256);
    const dim3 gEmb(BATCH * LSEQ), bEmb(256);

    // --- pre-swizzle all 6 conv weight tensors to f16 fragment order ---
    va_swizzle_w<<<gSwz, bSwz, 0, stream>>>(dp[0], w_dp1);
    va_swizzle_w<<<gSwz, bSwz, 0, stream>>>(dp[4], w_dp2);
    va_swizzle_w<<<gSwz, bSwz, 0, stream>>>(pp[0], w_pp1);
    va_swizzle_w<<<gSwz, bSwz, 0, stream>>>(pp[4], w_pp2);
    va_swizzle_w<<<gSwz, bSwz, 0, stream>>>(ep[0], w_ep1);
    va_swizzle_w<<<gSwz, bSwz, 0, stream>>>(ep[4], w_ep2);

    // --- duration predictor (on x) ---
    va_conv_ln<<<gConv, bConv, 0, stream>>>(x,  w_dp1, dp[1], dp[2], dp[3], y1);
    va_conv_ln<<<gConv, bConv, 0, stream>>>(y1, w_dp2, dp[5], dp[6], dp[7], y2);
    va_linear_mask<<<gLin, bLin, 0, stream>>>(y2, dp[8], dp[9], mask, outp + D_OFF);

    // --- pitch predictor (on x) ---
    va_conv_ln<<<gConv, bConv, 0, stream>>>(x,  w_pp1, pp[1], pp[2], pp[3], y1);
    va_conv_ln<<<gConv, bConv, 0, stream>>>(y1, w_pp2, pp[5], pp[6], pp[7], y2);
    va_linear_mask<<<gLin, bLin, 0, stream>>>(y2, pp[8], pp[9], mask, outp + P_OFF);

    // --- xe = x + pitch_emb[searchsorted(pitch_bins, pitch_target)] ---
    va_emb_add<<<gEmb, bEmb, 0, stream>>>(x, pitchT, pbins, pemb, xe);

    // --- energy predictor (on xe) ---
    va_conv_ln<<<gConv, bConv, 0, stream>>>(xe, w_ep1, ep[1], ep[2], ep[3], y1);
    va_conv_ln<<<gConv, bConv, 0, stream>>>(y1, w_ep2, ep[5], ep[6], ep[7], y2);
    va_linear_mask<<<gLin, bLin, 0, stream>>>(y2, ep[8], ep[9], mask, outp + E_OFF);

    // --- xe2 = xe + energy_emb[...]; reuse y1 (free after energy predictor) ---
    va_emb_add<<<gEmb, bEmb, 0, stream>>>(xe, energT, ebins, eemb, y1);

    // --- length regulate ---
    va_cumsum<<<dim3(BATCH), dim3(LSEQ), 0, stream>>>(dur, cum, outp + ML_OFF);
    va_length_reg<<<dim3(BATCH * TMAX), dim3(256), 0, stream>>>(y1, cum, outp + OUT_OFF);
}
